// SEALTargetAware_31782757991012
// MI455X (gfx1250) — matrix operand order repaired and checked by reading, not compiled, not run on hardware
//
#include <hip/hip_runtime.h>
#include <hip/hip_bf16.h>
#include <cstdint>

typedef _Float16 v16h __attribute__((ext_vector_type(16)));
typedef _Float16 v8h  __attribute__((ext_vector_type(8)));
typedef float    v8f  __attribute__((ext_vector_type(8)));
typedef int      v4i  __attribute__((ext_vector_type(4)));

#if defined(__AMDGCN__) && __has_builtin(__builtin_amdgcn_global_load_async_to_lds_b128) && \
    __has_builtin(__builtin_amdgcn_s_wait_asynccnt)
#define USE_ASYNC_LDS 1
typedef __attribute__((address_space(1))) v4i* gas_v4i_ptr;
typedef __attribute__((address_space(3))) v4i* las_v4i_ptr;
#else
#define USE_ASYNC_LDS 0
#endif

// ---------------------------------------------------------------------------
// elementwise / graph kernels
// ---------------------------------------------------------------------------
__global__ void k_zero_f32(float* __restrict__ p, int n) {
  int i = blockIdx.x * blockDim.x + threadIdx.x;
  if (i < n) p[i] = 0.0f;
}

__global__ void k_deg(const int* __restrict__ dst, float* __restrict__ deg, int E) {
  int e = blockIdx.x * blockDim.x + threadIdx.x;
  if (e < E) atomicAdd(&deg[dst[e]], 1.0f);
}

__global__ void k_dinv(const float* __restrict__ deg, float* __restrict__ dinv, int n) {
  int i = blockIdx.x * blockDim.x + threadIdx.x;
  if (i < n) dinv[i] = rsqrtf(deg[i] + 1.0f);
}

// z = concat(x, emb[drnl], zero-pad) -> f16, row-major N x 192 (K padded 160->192)
__global__ void k_prep_z16(const float* __restrict__ x, const float* __restrict__ emb,
                           const int* __restrict__ drnl, _Float16* __restrict__ A16, int N) {
  int idx = blockIdx.x * blockDim.x + threadIdx.x;
  if (idx >= N * 192) return;
  int i = idx / 192;
  int c = idx - i * 192;
  float v = 0.0f;
  if (c < 128)      v = x[(size_t)i * 128 + c];
  else if (c < 160) v = emb[(size_t)drnl[i] * 32 + (c - 128)];
  A16[idx] = (_Float16)v;
}

// Repack W (Kreal x 128, f32 row-major; zero-pad to Kpad) into WMMA B-fragment order:
// Bp[kk][wave][lane][j] = W[kk*32 + kb + (j&7) + (j>>3)*16][wave*16 + (lane&15)], kb = lane<16?0:8
__global__ void k_repack_w(const float* __restrict__ W, _Float16* __restrict__ Bp,
                           int Kpad, int Kreal) {
  int idx = blockIdx.x * blockDim.x + threadIdx.x;
  if (idx >= Kpad * 128) return;
  int j    = idx & 15;
  int lane = (idx >> 4) & 31;
  int w    = (idx >> 9) & 7;
  int kk   = idx >> 12;
  int n    = w * 16 + (lane & 15);
  int kb   = (lane < 16) ? 0 : 8;
  int k    = kk * 32 + kb + (j & 7) + ((j >> 3) << 4);
  Bp[idx]  = (k < Kreal) ? (_Float16)W[(size_t)k * 128 + n] : (_Float16)0.0f;
}

// ---------------------------------------------------------------------------
// WMMA GEMM: C[M x 128] = A16[M x K] * Wpacked[K x 128]
// M, K multiples of 16 / 64; block: 16 rows x 128 cols; 8 waves, one 16x16 tile
// each; K stepped by 64 (two v_wmma per staged LDS tile). No guards.
// ---------------------------------------------------------------------------
__global__ __launch_bounds__(256)
void k_wmma_gemm(const _Float16* __restrict__ A16, const _Float16* __restrict__ Bp,
                 float* __restrict__ C, int K) {
  __shared__ _Float16 sA[16][72];            // 16x64 tile, 144B row (16B-aligned, bank-skewed)
  const int row0 = blockIdx.x * 16;
  const int wave = threadIdx.x >> 5;
  const int lane = threadIdx.x & 31;
  const int m    = lane & 15;
  const int kA   = (lane < 16) ? 0 : 8;      // ISA 16-bit A fragment layout
  const int nkt  = K >> 6;

#if USE_ASYNC_LDS
  const int  r      = (threadIdx.x >> 3) & 15;  // rows 0..15 (threads 0..127)
  const int  seg    = threadIdx.x & 7;          // 16B segment within 128B row
  const bool loader = threadIdx.x < 128;
  const size_t gofs = (size_t)(row0 + r) * K + seg * 8;
#else
  const int  r      = threadIdx.x >> 4;         // 0..15
  const int  q      = threadIdx.x & 15;         // 8B chunk within row
  const size_t gofs = (size_t)(row0 + r) * K + q * 4;
#endif

  v8f acc = {};
  for (int kt = 0; kt < nkt; ++kt) {
    const int k0 = kt << 6;
#if USE_ASYNC_LDS
    if (loader) {
      gas_v4i_ptr gp = (gas_v4i_ptr)(uintptr_t)(A16 + gofs + k0);
      las_v4i_ptr lp = (las_v4i_ptr)(uint32_t)(uintptr_t)(&sA[r][seg * 8]);
      __builtin_amdgcn_global_load_async_to_lds_b128(gp, lp, 0, 0);
    }
    __builtin_amdgcn_s_wait_asynccnt(0);
#else
    uint2 av = *(const uint2*)(A16 + gofs + k0);
    *(uint2*)&sA[r][q * 4] = av;
#endif
    __syncthreads();

    // A fragments: lanes 0-15 hold M=lane, K 0-7 & 16-23; lanes 16-31 K 8-15 & 24-31
    v8h a0 = *(const v8h*)&sA[m][kA];
    v8h a1 = *(const v8h*)&sA[m][16 + kA];
    v8h a2 = *(const v8h*)&sA[m][32 + kA];
    v8h a3 = *(const v8h*)&sA[m][48 + kA];
    v16h af0, af1;
#pragma unroll
    for (int j = 0; j < 8; ++j) {
      af0[j] = a0[j]; af0[j + 8] = a1[j];
      af1[j] = a2[j]; af1[j + 8] = a3[j];
    }

    // B fragments: prepacked, 16 contiguous halves per lane (two b128 loads each)
    const _Float16* bp = Bp + (((size_t)(2 * kt) * 8 + wave) * 32 + lane) * 16;
    v16h bf0 = *(const v16h*)bp;
    v16h bf1 = *(const v16h*)(bp + 8 * 32 * 16);

    acc = __builtin_amdgcn_wmma_f32_16x16x32_f16(false, af0, false, bf0,
                                                 (short)0, acc, false, false);
    acc = __builtin_amdgcn_wmma_f32_16x16x32_f16(false, af1, false, bf1,
                                                 (short)0, acc, false, false);
    if (kt + 1 < nkt)
      __builtin_prefetch(A16 + gofs + k0 + 64, 0, 1);
    __syncthreads();
  }

  // C/D layout: VGPR j -> M = j (lanes 0-15) / M = 8+j (lanes 16-31), N = lane&15
  const int n  = (wave << 4) + m;
  const int mb = (lane < 16) ? 0 : 8;
#pragma unroll
  for (int j = 0; j < 8; ++j)
    C[(size_t)(row0 + mb + j) * 128 + n] = acc[j];
}

// ---------------------------------------------------------------------------
// GCN scatter stage
// ---------------------------------------------------------------------------
__global__ void k_init_agg(const float* __restrict__ hw, const float* __restrict__ dinv,
                           const float* __restrict__ b, float* __restrict__ agg, int N) {
  int idx = blockIdx.x * blockDim.x + threadIdx.x;
  if (idx >= N * 128) return;
  int i = idx >> 7, c = idx & 127;
  float di = dinv[i];
  agg[idx] = fmaf(hw[idx], di * di, b[c]);
}

__global__ __launch_bounds__(256)
void k_scatter(const float* __restrict__ hw, const int* __restrict__ src,
               const int* __restrict__ dst, const float* __restrict__ dinv,
               float* __restrict__ agg, int E) {
  int e    = (int)((blockIdx.x * (size_t)blockDim.x + threadIdx.x) >> 5);
  int lane = threadIdx.x & 31;
  if (e >= E) return;
  int s = src[e], d = dst[e];
  float nrm = dinv[s] * dinv[d];
  const float4 mv = ((const float4*)(hw + (size_t)s * 128))[lane];
  float* ad = agg + (size_t)d * 128 + lane * 4;
  atomicAdd(ad + 0, mv.x * nrm);
  atomicAdd(ad + 1, mv.y * nrm);
  atomicAdd(ad + 2, mv.z * nrm);
  atomicAdd(ad + 3, mv.w * nrm);
}

__global__ void k_relu_cvt(float* __restrict__ a, _Float16* __restrict__ a16, int n) {
  int idx = blockIdx.x * blockDim.x + threadIdx.x;
  if (idx >= n) return;
  float v = fmaxf(a[idx], 0.0f);
  a[idx] = v;
  a16[idx] = (_Float16)v;
}

__global__ void k_relu(float* __restrict__ a, int n) {
  int idx = blockIdx.x * blockDim.x + threadIdx.x;
  if (idx < n) a[idx] = fmaxf(a[idx], 0.0f);
}

// ---------------------------------------------------------------------------
// readout: pool + pairwise features + MLP tail
// ---------------------------------------------------------------------------
__global__ void k_pool(const float* __restrict__ z, _Float16* __restrict__ f16, int NPG) {
  int g = blockIdx.x, c = threadIdx.x;
  const float* p = z + ((size_t)g * NPG) * 128 + c;
  float mx = -3.0e38f;
  for (int r = 0; r < NPG; ++r) mx = fmaxf(mx, p[(size_t)r * 128]);
  f16[(size_t)g * 640 + 512 + c] = (_Float16)mx;
}

__global__ void k_feats(const float* __restrict__ z, const int* __restrict__ tl,
                        const int* __restrict__ ptr, _Float16* __restrict__ f16) {
  int g = blockIdx.x, c = threadIdx.x;
  int u = ptr[g] + tl[2 * g];
  int v = ptr[g] + tl[2 * g + 1];
  float a = z[(size_t)u * 128 + c];
  float b = z[(size_t)v * 128 + c];
  _Float16* f = f16 + (size_t)g * 640;
  f[c]       = (_Float16)a;
  f[128 + c] = (_Float16)b;
  f[256 + c] = (_Float16)fabsf(a - b);
  f[384 + c] = (_Float16)(a * b);
}

__global__ void k_bias_relu(float* __restrict__ h, const float* __restrict__ b, int n) {
  int idx = blockIdx.x * blockDim.x + threadIdx.x;
  if (idx < n) h[idx] = fmaxf(h[idx] + b[idx & 127], 0.0f);
}

__global__ void k_mlp2(const float* __restrict__ h, const float* __restrict__ W,
                       const float* __restrict__ b, float* __restrict__ out, int G) {
  int i = blockIdx.x * blockDim.x + threadIdx.x;
  if (i >= 2 * G) return;
  int g = i >> 1, j = i & 1;
  float acc = b[j];
  const float* hp = h + (size_t)g * 128;
#pragma unroll 4
  for (int c = 0; c < 128; ++c) acc = fmaf(hp[c], W[c * 2 + j], acc);
  out[i] = acc;
}

// ---------------------------------------------------------------------------
// launcher
// ---------------------------------------------------------------------------
extern "C" void kernel_launch(void* const* d_in, const int* in_sizes, int n_in,
                              void* d_out, int out_size, void* d_ws, size_t ws_size,
                              hipStream_t stream) {
  const float* x   = (const float*)d_in[0];
  const float* emb = (const float*)d_in[1];
  const float* W1  = (const float*)d_in[2];
  const float* b1  = (const float*)d_in[3];
  const float* W2  = (const float*)d_in[4];
  const float* b2  = (const float*)d_in[5];
  const float* Wm1 = (const float*)d_in[6];
  const float* bm1 = (const float*)d_in[7];
  const float* Wm2 = (const float*)d_in[8];
  const float* bm2 = (const float*)d_in[9];
  const int* edge_index   = (const int*)d_in[10];
  const int* drnl         = (const int*)d_in[11];
  const int* target_local = (const int*)d_in[13];
  const int* ptr          = (const int*)d_in[14];

  const int N    = in_sizes[11];           // 100000 (multiple of 16)
  const int E    = in_sizes[10] / 2;       // 1600000
  const int G    = in_sizes[14] - 1;       // 1000
  const int NPG  = N / G;                  // 100
  const int Gpad = (G + 15) & ~15;         // 1008
  const int* e_src = edge_index;
  const int* e_dst = edge_index + E;

  // workspace layout
  char* ws = (char*)d_ws;
  size_t off = 0;
  auto take = [&](size_t bytes) -> char* {
    char* p = ws + off;
    off = (off + bytes + 255) & ~(size_t)255;
    return p;
  };
  float*    hw    = (float*)take((size_t)N * 128 * sizeof(float));
  float*    agg   = (float*)take((size_t)N * 128 * sizeof(float));
  _Float16* A16   = (_Float16*)take((size_t)N * 192 * sizeof(_Float16));
  float*    deg   = (float*)take((size_t)N * sizeof(float));
  float*    dinv  = (float*)take((size_t)N * sizeof(float));
  _Float16* Bp    = (_Float16*)take((size_t)640 * 128 * sizeof(_Float16));
  _Float16* fts16 = (_Float16*)take((size_t)Gpad * 640 * sizeof(_Float16));
  float*    hidm  = (float*)take((size_t)Gpad * 128 * sizeof(float));
  (void)ws_size;

  auto cdiv = [](long long a, long long b) -> unsigned { return (unsigned)((a + b - 1) / b); };
  const int T = 256;

  // degree / normalization
  k_zero_f32<<<cdiv(N, T), T, 0, stream>>>(deg, N);
  k_deg<<<cdiv(E, T), T, 0, stream>>>(e_dst, deg, E);
  k_dinv<<<cdiv(N, T), T, 0, stream>>>(deg, dinv, N);

  // z = [x | emb[drnl] | 0-pad] as f16 (K = 192)
  k_prep_z16<<<cdiv((long long)N * 192, T), T, 0, stream>>>(x, emb, drnl, A16, N);

  // ---- layer 1: hw = z @ W1 ; agg = scatter + self ; relu ----
  k_repack_w<<<cdiv(192 * 128, T), T, 0, stream>>>(W1, Bp, 192, 160);
  k_wmma_gemm<<<N / 16, T, 0, stream>>>(A16, Bp, hw, 192);
  k_init_agg<<<cdiv((long long)N * 128, T), T, 0, stream>>>(hw, dinv, b1, agg, N);
  k_scatter<<<cdiv((long long)E * 32, T), T, 0, stream>>>(hw, e_src, e_dst, dinv, agg, E);
  k_relu_cvt<<<cdiv((long long)N * 128, T), T, 0, stream>>>(agg, A16, N * 128);

  // ---- layer 2 ----
  k_repack_w<<<cdiv(128 * 128, T), T, 0, stream>>>(W2, Bp, 128, 128);
  k_wmma_gemm<<<N / 16, T, 0, stream>>>(A16, Bp, hw, 128);
  k_init_agg<<<cdiv((long long)N * 128, T), T, 0, stream>>>(hw, dinv, b2, agg, N);
  k_scatter<<<cdiv((long long)E * 32, T), T, 0, stream>>>(hw, e_src, e_dst, dinv, agg, E);
  k_relu<<<cdiv((long long)N * 128, T), T, 0, stream>>>(agg, N * 128);   // agg == z2

  // ---- readout ----
  k_zero_f32<<<cdiv((long long)Gpad * 640 / 2, T), T, 0, stream>>>((float*)fts16, Gpad * 640 / 2);
  k_pool<<<G, 128, 0, stream>>>(agg, fts16, NPG);
  k_feats<<<G, 128, 0, stream>>>(agg, target_local, ptr, fts16);

  k_repack_w<<<cdiv(640 * 128, T), T, 0, stream>>>(Wm1, Bp, 640, 640);
  k_wmma_gemm<<<Gpad / 16, T, 0, stream>>>(fts16, Bp, hidm, 640);
  k_bias_relu<<<cdiv((long long)G * 128, T), T, 0, stream>>>(hidm, bm1, G * 128);

  k_mlp2<<<cdiv(2 * G, T), T, 0, stream>>>(hidm, Wm2, bm2, (float*)d_out, G);
}
